// Memory_32220844655352
// MI455X (gfx1250) — matrix operand order repaired and checked by reading, not compiled, not run on hardware
//
#include <hip/hip_runtime.h>
#include <math.h>

// Problem constants (from reference)
#define B_    256
#define H_    6
#define N_    2048
#define M_    64
#define DIN_  128
#define DW_   12288          // H*N (flattened w part of x)
#define DCAT_ 12416          // DW_ + DIN_
#define DOUT_ 384            // H*M
#define NC_   1152           // 3*DOUT_ (Wk|We|Wa fused)
#define KCHUNKS_ 388         // DCAT_/32
#define KSPLIT_  8

typedef __attribute__((ext_vector_type(2)))  float  v2f;
typedef __attribute__((ext_vector_type(8)))  float  v8f;
typedef __attribute__((ext_vector_type(16))) __bf16 v16bf;
typedef __attribute__((ext_vector_type(4)))  unsigned int v4u;
typedef __attribute__((ext_vector_type(4)))  int    v4i_;
typedef __attribute__((ext_vector_type(8)))  int    v8i_;

// Workspace layout (float elements)
#define WS_PARTIAL 0
#define WS_KNORM  (KSPLIT_ * B_ * NC_)            // 2,359,296
#define WS_E      (WS_KNORM + B_ * DOUT_)
#define WS_A      (WS_E + B_ * DOUT_)
#define WS_LOGITS (WS_A + B_ * DOUT_)
// total = WS_LOGITS + B_*H_*N_ = 5,799,936 floats (~23.2 MB)

// ---------------------------------------------------------------------------
// TDM: 1-row tensor tile (width elems of 4B) -> LDS, per CDNA5 D# layout.
// Group0: count=1 | lds_addr | global_addr[56:0] | type=2
// Group1: data_size=4B | tensor_dim0 | tensor_dim1=1 | tile_dim0 | tile_dim1=1
//         | tensor_dim0_stride
// Low 32 bits of a flat LDS pointer are the LDS byte offset (aperture in high
// bits), so lds_off comes from a plain pointer truncation.
// ---------------------------------------------------------------------------
__device__ __forceinline__ void tdm_load_row_to_lds(unsigned int lds_off,
                                                    const void* gptr,
                                                    unsigned int width_elems,
                                                    unsigned int stride_elems)
{
    const unsigned long long ga = (unsigned long long)(size_t)gptr;
    v4u g0;
    g0[0] = 1u;                                                  // count=1
    g0[1] = lds_off;                                             // LDS byte addr
    g0[2] = (unsigned int)ga;                                    // addr[31:0]
    g0[3] = (unsigned int)((ga >> 32) & 0x01FFFFFFull) | (2u << 30); // addr[56:32] | type=2

    v8i_ g1;
    g1[0] = (int)(2u << 16);                                     // data_size=4B, mask=0
    g1[1] = (int)((width_elems & 0xFFFFu) << 16);                // tensor_dim0[15:0] @ bit48
    g1[2] = (int)(((width_elems >> 16) & 0xFFFFu) | (1u << 16)); // dim0 hi | tensor_dim1=1
    g1[3] = (int)((width_elems & 0xFFFFu) << 16);                // tile_dim0 @ bits127:112
    g1[4] = 1;                                                   // tile_dim1=1, tile_dim2=0
    g1[5] = (int)stride_elems;                                   // tensor_dim0_stride lo32
    g1[6] = 0;
    g1[7] = 0;

    const v4i_ z4 = {0, 0, 0, 0};
#if __clang_major__ >= 23
    const v8i_ z8 = {0, 0, 0, 0, 0, 0, 0, 0};
    __builtin_amdgcn_tensor_load_to_lds(g0, g1, z4, z4, z8, 0);
#else
    __builtin_amdgcn_tensor_load_to_lds(g0, g1, z4, z4, 0);
#endif
}

// ---------------------------------------------------------------------------
// Kernel 1: Y_partial[z] = X(tile) @ [Wk|We|Wa](tile)  via bf16 WMMA, fp32 acc
// X[row][k] = (k < 12288) ? w[row*12288+k] : inputs[row*128+k-12288]
// Grid (2, 9, KSPLIT), block 256 (8 waves). WG tile: 128 rows x 128 cols.
// ---------------------------------------------------------------------------
__global__ __launch_bounds__(256)
void gemm_xw_kernel(const float* __restrict__ w, const float* __restrict__ inp,
                    const float* __restrict__ Wk, const float* __restrict__ We,
                    const float* __restrict__ Wa, float* __restrict__ partial)
{
    // W tile staged transposed [n][k] as bf16; row stride 48 (96B) keeps the
    // two 16-elem K-halves 32B-aligned and spreads LDS bank starts (24n mod 64).
    __shared__ __align__(32) __bf16 ldsb[128 * 48];

    const int tid  = threadIdx.x;
    const int wave = tid >> 5;
    const int lane = tid & 31;
    const int lrow = lane & 15;     // matrix row (A) / col (B,D) owned by lane
    const int lhi  = lane >> 4;     // half-wave select

    const int rbase = blockIdx.x * 128;      // X row tile base
    const int nbase = blockIdx.y * 128;      // fused output col tile base
    const int z     = blockIdx.z;

    // 128-wide col tile lies fully inside one of Wk/We/Wa (384 = 3*128)
    const int which = nbase / DOUT_;
    const float* Wsrc = (which == 0) ? Wk : (which == 1) ? We : Wa;
    const int ncol0 = nbase - which * DOUT_;

    const int row = rbase + wave * 16 + lrow;    // A-matrix row for this lane
    const float* wrow = w   + (size_t)row * DW_;
    const float* irow = inp + (size_t)row * DIN_;

    int c0 = z * 49;
    int c1 = c0 + 49; if (c1 > KCHUNKS_) c1 = KCHUNKS_;

    v8f acc[8];
    #pragma unroll
    for (int t = 0; t < 8; ++t) acc[t] = (v8f){0.f,0.f,0.f,0.f,0.f,0.f,0.f,0.f};

    // Staging map: thread covers 4 consecutive n (float4) x 4 consecutive k.
    const int tc4 = (tid & 31) * 4;   // n offset: lanes -> 512B contiguous b128
    const int tr2 = tid >> 5;         // k quarter-group (0..7)

    // Thread-fixed base; per-chunk advance is one pointer add; the i*DOUT_
    // steps fold into the 24-bit instruction offset (i*1536 bytes).
    const float* wp = Wsrc + (size_t)(c0 * 32 + tr2 * 4) * DOUT_ + ncol0 + tc4;

    for (int c = c0; c < c1; ++c) {
        const int k0 = c * 32;

        // ---- stage W 32x128 fp32 -> bf16 transposed in LDS ----
        #pragma unroll
        for (int i = 0; i < 4; ++i) {
            const float4 u = *(const float4*)(wp + i * DOUT_);
            const int kl = tr2 * 4 + i;
            ldsb[(tc4 + 0) * 48 + kl] = (__bf16)u.x;
            ldsb[(tc4 + 1) * 48 + kl] = (__bf16)u.y;
            ldsb[(tc4 + 2) * 48 + kl] = (__bf16)u.z;
            ldsb[(tc4 + 3) * 48 + kl] = (__bf16)u.w;
        }
        if (c + 1 < c1)   // warm next tile (global_prefetch_b8)
            __builtin_prefetch(wp + 32 * DOUT_, 0, 1);
        wp += 32 * DOUT_;
        __syncthreads();

        // ---- A fragment: 16 consecutive K per lane half:
        // lane<16 holds K=[0,16); lane>=16 holds K=[16,32) ----
        v16bf af;
        {
            const int ka0 = k0 + lhi * 16;
            const int ka1 = k0 + lhi * 16 + 8;
            const float* p0 = (ka0 < DW_) ? (wrow + ka0) : (irow + (ka0 - DW_));
            const float* p1 = (ka1 < DW_) ? (wrow + ka1) : (irow + (ka1 - DW_));
            const float4 u0 = ((const float4*)p0)[0];
            const float4 u1 = ((const float4*)p0)[1];
            const float4 u2 = ((const float4*)p1)[0];
            const float4 u3 = ((const float4*)p1)[1];
            af[0]=(__bf16)u0.x; af[1]=(__bf16)u0.y; af[2]=(__bf16)u0.z; af[3]=(__bf16)u0.w;
            af[4]=(__bf16)u1.x; af[5]=(__bf16)u1.y; af[6]=(__bf16)u1.z; af[7]=(__bf16)u1.w;
            af[8]=(__bf16)u2.x; af[9]=(__bf16)u2.y; af[10]=(__bf16)u2.z; af[11]=(__bf16)u2.w;
            af[12]=(__bf16)u3.x; af[13]=(__bf16)u3.y; af[14]=(__bf16)u3.z; af[15]=(__bf16)u3.w;
        }

        // ---- 8 column sub-tiles: B frag = 16 consecutive K (lhi*16) for col ----
        #pragma unroll
        for (int nt = 0; nt < 8; ++nt) {
            const int nl = nt * 16 + lrow;
            const v16bf bf = *(const v16bf*)&ldsb[nl * 48 + lhi * 16];
            acc[nt] = __builtin_amdgcn_wmma_f32_16x16x32_bf16(
                false, af, false, bf, (short)0, acc[nt], false, false);
        }
        __syncthreads();
    }

    // D layout: VGPR v, lane<16 -> row v, lane>=16 -> row v+8; col = lrow
    float* out = partial + (size_t)z * B_ * NC_;
    #pragma unroll
    for (int nt = 0; nt < 8; ++nt) {
        #pragma unroll
        for (int v = 0; v < 8; ++v) {
            const int r  = rbase + wave * 16 + v + lhi * 8;
            const int cn = nbase + nt * 16 + lrow;
            out[(size_t)r * NC_ + cn] = acc[nt][v];
        }
    }
}

// ---------------------------------------------------------------------------
// Kernel 2: reduce split-K partials + bias; k_norm per head; sigmoid(e), sigmoid(a)
// Grid (B), block 384 (thread j covers column j of each of Wk/We/Wa outputs).
// ---------------------------------------------------------------------------
__global__ __launch_bounds__(384)
void heads_kernel(const float* __restrict__ partial,
                  const float* __restrict__ bk, const float* __restrict__ be,
                  const float* __restrict__ ba,
                  float* __restrict__ knorm, float* __restrict__ ev,
                  float* __restrict__ av)
{
    __shared__ float sh[DOUT_];
    const int b = blockIdx.x, j = threadIdx.x;

    float yk = bk[j], ye = be[j], ya = ba[j];
    #pragma unroll
    for (int z = 0; z < KSPLIT_; ++z) {
        const float* p = partial + ((size_t)z * B_ + b) * NC_;
        yk += p[j]; ye += p[DOUT_ + j]; ya += p[2 * DOUT_ + j];
    }
    ev[b * DOUT_ + j] = 1.f / (1.f + expf(-ye));
    av[b * DOUT_ + j] = 1.f / (1.f + expf(-ya));

    sh[j] = yk * yk;
    __syncthreads();
    #pragma unroll
    for (int s = 32; s > 0; s >>= 1) {          // reduce within 64-wide head group
        if ((j & 63) < s) sh[j] += sh[j + s];
        __syncthreads();
    }
    knorm[b * DOUT_ + j] = yk * rsqrtf(sh[j & ~63]);
}

// ---------------------------------------------------------------------------
// Kernel 3: logits[b,h,n] = k_norm[b,h,:] . (memory[b,n,:]/||memory[b,n,:]||)
// Grid (N/256, B), block 256, one memory row per thread (float4 loads).
// k_norm block (384 floats) staged into LDS by the Tensor Data Mover.
// ---------------------------------------------------------------------------
__global__ __launch_bounds__(256)
void logits_kernel(const float* __restrict__ memory,
                   const float* __restrict__ knorm,
                   float* __restrict__ logits)
{
    __shared__ float sk[DOUT_];
    const int b = blockIdx.y;
    const int n = blockIdx.x * 256 + threadIdx.x;

    if (threadIdx.x < 32) {              // wave0 issues the TDM DMA (EXEC ignored)
        tdm_load_row_to_lds((unsigned int)(size_t)(void*)sk,
                            knorm + (size_t)b * DOUT_, DOUT_, DOUT_);
        __builtin_amdgcn_s_wait_tensorcnt(0);
    }
    __syncthreads();

    const float4* mrow = (const float4*)(memory + ((size_t)b * N_ + n) * M_);
    float ss = 0.f;
    float dot[H_] = {0.f, 0.f, 0.f, 0.f, 0.f, 0.f};
    #pragma unroll
    for (int q = 0; q < 16; ++q) {
        const float4 v = mrow[q];
        ss += v.x * v.x + v.y * v.y + v.z * v.z + v.w * v.w;
        #pragma unroll
        for (int h = 0; h < H_; ++h) {
            const float* kp = &sk[h * 64 + q * 4];
            dot[h] += v.x * kp[0] + v.y * kp[1] + v.z * kp[2] + v.w * kp[3];
        }
    }
    const float inv = rsqrtf(ss);
    #pragma unroll
    for (int h = 0; h < H_; ++h)
        logits[((size_t)(b * H_ + h) << 11) + n] = dot[h] * inv;
}

// ---------------------------------------------------------------------------
// Kernel 4: softmax over n (2048) per (b,h) row -> w_new (directly into d_out)
// Grid (B*H), block 256, 8 elements/thread.
// ---------------------------------------------------------------------------
__global__ __launch_bounds__(256)
void softmax_kernel(const float* __restrict__ logits, float* __restrict__ wnew)
{
    __shared__ float sh[256];
    const int row = blockIdx.x, tid = threadIdx.x;
    const float* src = logits + (size_t)row * N_;

    float v[8];
    float mx = -3.402823466e38f;
    #pragma unroll
    for (int i = 0; i < 8; ++i) { v[i] = src[tid + i * 256]; mx = fmaxf(mx, v[i]); }
    sh[tid] = mx; __syncthreads();
    #pragma unroll
    for (int s = 128; s > 0; s >>= 1) {
        if (tid < s) sh[tid] = fmaxf(sh[tid], sh[tid + s]);
        __syncthreads();
    }
    mx = sh[0]; __syncthreads();

    float sum = 0.f;
    #pragma unroll
    for (int i = 0; i < 8; ++i) { v[i] = expf(v[i] - mx); sum += v[i]; }
    sh[tid] = sum; __syncthreads();
    #pragma unroll
    for (int s = 128; s > 0; s >>= 1) {
        if (tid < s) sh[tid] += sh[tid + s];
        __syncthreads();
    }
    const float inv = 1.f / sh[0];

    float* dst = wnew + (size_t)row * N_;
    #pragma unroll
    for (int i = 0; i < 8; ++i) dst[tid + i * 256] = v[i] * inv;
}

// ---------------------------------------------------------------------------
// Kernel 5: erase/add = w_new^T[16n x 8k(h-pad)] @ e/a[8k x 16m] via f32 WMMA
// (two chained K=4 WMMAs, h=6,7 zero-padded), then fused memory update:
// new_mem = mem - mem^2*erase + add.   Grid (N/128, B), block 256 (8 waves).
// ---------------------------------------------------------------------------
__global__ __launch_bounds__(256)
void update_kernel(const float* __restrict__ memory,
                   const float* __restrict__ wnew,
                   const float* __restrict__ ev, const float* __restrict__ av,
                   float* __restrict__ outmem)
{
    const int tid  = threadIdx.x;
    const int wave = tid >> 5;
    const int lane = tid & 31;
    const int lrow = lane & 15;
    const int lhi  = lane >> 4;
    const int b  = blockIdx.y;
    const int n0 = blockIdx.x * 128 + wave * 16;

    // A frag (f32 16x4): lane<16 holds K=0,1 ; lane>=16 holds K=2,3  (K == head)
    const float* wn = wnew + (size_t)b * H_ * N_;
    const int nA = n0 + lrow;
    v2f alo, ahi;
    alo.x = wn[(lhi * 2 + 0) * N_ + nA];            // h = 0/2
    alo.y = wn[(lhi * 2 + 1) * N_ + nA];            // h = 1/3
    ahi.x = lhi ? 0.f : wn[4 * N_ + nA];            // h = 4 (h=6 pad -> 0)
    ahi.y = lhi ? 0.f : wn[5 * N_ + nA];            // h = 5 (h=7 pad -> 0)

    const float* ep  = ev + b * DOUT_;
    const float* ap  = av + b * DOUT_;
    const float* mem = memory + ((size_t)b * N_ + n0) * M_;
    float*       om  = outmem + ((size_t)b * N_ + n0) * M_;

    #pragma unroll
    for (int mt = 0; mt < 4; ++mt) {
        const int mc = mt * 16 + lrow;

        // B frag (f32 4x16): lane<16 holds K=0,1 ; lane>=16 holds K=2,3
        v2f blo, bhi;
        blo.x = ep[(lhi * 2 + 0) * 64 + mc];
        blo.y = ep[(lhi * 2 + 1) * 64 + mc];
        bhi.x = lhi ? 0.f : ep[4 * 64 + mc];
        bhi.y = lhi ? 0.f : ep[5 * 64 + mc];
        v8f er = {0.f,0.f,0.f,0.f,0.f,0.f,0.f,0.f};
        er = __builtin_amdgcn_wmma_f32_16x16x4_f32(false, alo, false, blo,
                                                   (short)0, er, false, false);
        er = __builtin_amdgcn_wmma_f32_16x16x4_f32(false, ahi, false, bhi,
                                                   (short)0, er, false, false);

        blo.x = ap[(lhi * 2 + 0) * 64 + mc];
        blo.y = ap[(lhi * 2 + 1) * 64 + mc];
        bhi.x = lhi ? 0.f : ap[4 * 64 + mc];
        bhi.y = lhi ? 0.f : ap[5 * 64 + mc];
        v8f ad = {0.f,0.f,0.f,0.f,0.f,0.f,0.f,0.f};
        ad = __builtin_amdgcn_wmma_f32_16x16x4_f32(false, alo, false, blo,
                                                   (short)0, ad, false, false);
        ad = __builtin_amdgcn_wmma_f32_16x16x4_f32(false, ahi, false, bhi,
                                                   (short)0, ad, false, false);

        // D layout: VGPR v -> row(n) = v + lhi*8 ; col(m) = mc. Fused update.
        #pragma unroll
        for (int v = 0; v < 8; ++v) {
            const int idx = (v + lhi * 8) * M_ + mc;
            const float m0 = mem[idx];
            om[idx] = m0 - m0 * m0 * er[v] + ad[v];
        }
    }
}

// ---------------------------------------------------------------------------
extern "C" void kernel_launch(void* const* d_in, const int* in_sizes, int n_in,
                              void* d_out, int out_size, void* d_ws, size_t ws_size,
                              hipStream_t stream)
{
    const float* inputs = (const float*)d_in[0];   // [B, DIN]
    const float* memory = (const float*)d_in[1];   // [B, N, M]
    const float* w      = (const float*)d_in[2];   // [B, H, N]
    const float* Wk     = (const float*)d_in[3];   // [DCAT, DOUT]
    const float* bk     = (const float*)d_in[4];
    const float* We     = (const float*)d_in[5];
    const float* be     = (const float*)d_in[6];
    const float* Wa     = (const float*)d_in[7];
    const float* ba     = (const float*)d_in[8];

    float* out      = (float*)d_out;                      // new_memory [B,N,M]
    float* wnew_out = out + (size_t)B_ * N_ * M_;         // w_new [B,H,N]

    float* ws      = (float*)d_ws;
    float* partial = ws + WS_PARTIAL;
    float* knorm   = ws + WS_KNORM;
    float* ev      = ws + WS_E;
    float* av      = ws + WS_A;
    float* logits  = ws + WS_LOGITS;

    gemm_xw_kernel<<<dim3(2, 9, KSPLIT_), 256, 0, stream>>>(w, inputs, Wk, We, Wa, partial);
    heads_kernel<<<dim3(B_), 384, 0, stream>>>(partial, bk, be, ba, knorm, ev, av);
    logits_kernel<<<dim3(N_ / 256, B_), 256, 0, stream>>>(memory, knorm, logits);
    softmax_kernel<<<dim3(B_ * H_), 256, 0, stream>>>(logits, wnew_out);
    update_kernel<<<dim3(N_ / 128, B_), 256, 0, stream>>>(memory, wnew_out, ev, av, out);
}
